// Encoder_25752623907305
// MI455X (gfx1250) — compile-verified, matrix operand order checked by reading, not run on hardware
//
#include <hip/hip_runtime.h>
#include <hip/hip_fp16.h>

// D (feature dim) is 128 throughout; L layers; output = concat(xpool[B*D], x[N*D]).

typedef __attribute__((ext_vector_type(16))) _Float16 v16h;
typedef __attribute__((ext_vector_type(8)))  float    v8f;

#define FD 128
#define BN_EPS 1e-4f
#define BN_ROWS 256

// ---------------------------------------------------------------- utilities
__global__ void fill_kernel(float* __restrict__ p, float v, int n) {
  int t = blockIdx.x * blockDim.x + threadIdx.x;
  if (t < n) p[t] = v;
}

__global__ void deg_count_kernel(const int* __restrict__ dst, float* __restrict__ deg, int E) {
  int t = blockIdx.x * blockDim.x + threadIdx.x;
  if (t < E) atomicAdd(&deg[dst[t]], 1.0f);
}

__global__ void dinv_kernel(float* __restrict__ deg, int n) {
  int t = blockIdx.x * blockDim.x + threadIdx.x;
  if (t < n) deg[t] = rsqrtf(deg[t]);   // deg >= 1 (self loop), matches where(deg>0,...)
}

// ------------------------------------------------- pack W into WMMA B layout
// B-matrix (32x16, KxN) f16 lane layout: lane l holds column n = l&15,
// K = kbase + i (i<8) and K = 16 + kbase + (i-8) (i>=8), kbase = (l>>4)*8.
// Packed as [ntile(8)][kchunk(4)][lane(32)][16 halves] -> contiguous 32B/lane.
__global__ void packW_kernel(const float* __restrict__ W, _Float16* __restrict__ Wh) {
  int idx = blockIdx.x * blockDim.x + threadIdx.x;      // 8*4*32*16 = 16384
  if (idx >= 8 * 4 * 32 * 16) return;
  int i     = idx & 15;
  int lane  = (idx >> 4) & 31;
  int kc    = (idx >> 9) & 3;
  int ntile = (idx >> 11);
  int n     = ntile * 16 + (lane & 15);
  int kbase = (lane >> 4) * 8;
  int k     = kc * 32 + ((i < 8) ? (kbase + i) : (16 + kbase + (i - 8)));
  Wh[idx] = (_Float16)W[k * FD + n];
}

// ------------------------------------------------------------- WMMA GEMM
// H[N,128] = X[N,128] @ W[128,128].
// One wave -> two 16x16 tiles of H (32 rows) sharing one B fragment:
// two independent accumulators fill the WMMA->VALU hazard window and halve
// B-matrix loads. Block = 8 waves = one 32-row stripe, wave w = column tile w.
__global__ __launch_bounds__(256) void gemm_wmma_kernel(const float* __restrict__ X,
                                                        const _Float16* __restrict__ Wh,
                                                        float* __restrict__ H, int Nrows) {
  const int lane  = threadIdx.x & 31;
  const int wave  = threadIdx.x >> 5;      // ntile 0..7
  const int mtile = blockIdx.x;            // 32-row stripe

  int mrow0 = mtile * 32 + (lane & 15);
  int mrow1 = mrow0 + 16;
  if (mrow0 >= Nrows) mrow0 = Nrows - 1;   // clamp, no EXEC divergence
  if (mrow1 >= Nrows) mrow1 = Nrows - 1;
  const int kbase = (lane >> 4) * 8;

  v8f acc0 = {};
  v8f acc1 = {};
  #pragma unroll
  for (int kc = 0; kc < 4; ++kc) {
    const float* xr0 = X + (size_t)mrow0 * FD + kc * 32 + kbase;
    const float* xr1 = X + (size_t)mrow1 * FD + kc * 32 + kbase;
    v16h a0, a1;
    #pragma unroll
    for (int i = 0; i < 8; ++i) {
      a0[i]     = (_Float16)xr0[i];        // K = kc*32 + kbase + i
      a0[i + 8] = (_Float16)xr0[16 + i];   // K = kc*32 + 16 + kbase + i
      a1[i]     = (_Float16)xr1[i];
      a1[i + 8] = (_Float16)xr1[16 + i];
    }
    v16h b = *(const v16h*)(Wh + ((size_t)(wave * 4 + kc) * 32 + lane) * 16);
    acc0 = __builtin_amdgcn_wmma_f32_16x16x32_f16(false, a0, false, b,
                                                  (short)0, acc0, false, false);
    acc1 = __builtin_amdgcn_wmma_f32_16x16x32_f16(false, a1, false, b,
                                                  (short)0, acc1, false, false);
  }

  // C/D layout: lane l -> N = (l&15), VGPR j -> M = j + (l>>4)*8
  const int mbase0 = mtile * 32 + (lane >> 4) * 8;
  const int mbase1 = mbase0 + 16;
  const int ncol   = wave * 16 + (lane & 15);
  #pragma unroll
  for (int j = 0; j < 8; ++j) {
    int m0 = mbase0 + j;
    if (m0 < Nrows) H[(size_t)m0 * FD + ncol] = acc0[j];
  }
  #pragma unroll
  for (int j = 0; j < 8; ++j) {
    int m1 = mbase1 + j;
    if (m1 < Nrows) H[(size_t)m1 * FD + ncol] = acc1[j];
  }
}

// --------------------------------------------- agg = bias + dinv^2 * h (self loop)
__global__ void agg_init_kernel(const float* __restrict__ H, const float* __restrict__ dinv,
                                const float* __restrict__ bias, float* __restrict__ agg, int N) {
  int t = blockIdx.x * blockDim.x + threadIdx.x;
  int n = t >> 5;
  if (n >= N) return;
  int f = (t & 31) * 4;
  float w = dinv[n]; w = w * w;
  float4 h = *(const float4*)(H + (size_t)n * FD + f);
  float4 b = *(const float4*)(bias + f);
  float4 o = { h.x * w + b.x, h.y * w + b.y, h.z * w + b.z, h.w * w + b.w };
  *(float4*)(agg + (size_t)n * FD + f) = o;
}

// --------------------------------------------- agg[dst] += h[src] * dinv[s]*dinv[d]
__global__ void scatter_kernel(const int* __restrict__ src, const int* __restrict__ dst,
                               const float* __restrict__ dinv, const float* __restrict__ H,
                               float* __restrict__ agg, int E) {
  int t = blockIdx.x * blockDim.x + threadIdx.x;
  int e = t >> 5;
  if (e >= E) return;
  int f = (t & 31) * 4;
  int s = src[e], d = dst[e];
  float nrm = dinv[s] * dinv[d];
  float4 h = *(const float4*)(H + (size_t)s * FD + f);
  float* o = agg + (size_t)d * FD + f;
  atomicAdd(o + 0, h.x * nrm);
  atomicAdd(o + 1, h.y * nrm);
  atomicAdd(o + 2, h.z * nrm);
  atomicAdd(o + 3, h.w * nrm);
}

// --------------------------------------------- a = sigmoid(agg) (in place) + BN partials
__global__ __launch_bounds__(128) void bn_stats_kernel(float* __restrict__ agg,
                                                       float* __restrict__ sums, int N) {
  const int d  = threadIdx.x;              // feature
  const int n0 = blockIdx.x * BN_ROWS;
  float s = 0.f, ss = 0.f;
  for (int r = 0; r < BN_ROWS; ++r) {
    int n = n0 + r;
    if (n >= N) break;
    float v = agg[(size_t)n * FD + d];
    float a = 1.0f / (1.0f + __expf(-v));
    agg[(size_t)n * FD + d] = a;
    s  += a;
    ss += a * a;
  }
  atomicAdd(&sums[d], s);
  atomicAdd(&sums[FD + d], ss);
}

// sums -> (scale, shift) in place
__global__ void bn_finalize_kernel(float* __restrict__ sums, const float* __restrict__ gamma,
                                   const float* __restrict__ beta, int N) {
  int d = threadIdx.x;
  if (d >= FD) return;
  float invN = 1.0f / (float)N;
  float m   = sums[d] * invN;
  float var = sums[FD + d] * invN - m * m;     // biased variance
  float sc  = gamma[d] * rsqrtf(var + BN_EPS);
  sums[d]      = sc;
  sums[FD + d] = beta[d] - m * sc;
}

__global__ void bn_apply_kernel(const float* __restrict__ agg, const float* __restrict__ ss,
                                float* __restrict__ xo, int N) {
  int t = blockIdx.x * blockDim.x + threadIdx.x;
  int n = t >> 5;
  if (n >= N) return;
  int f = (t & 31) * 4;
  float4 a = *(const float4*)(agg + (size_t)n * FD + f);
  float4 o = { a.x * ss[f + 0] + ss[FD + f + 0],
               a.y * ss[f + 1] + ss[FD + f + 1],
               a.z * ss[f + 2] + ss[FD + f + 2],
               a.w * ss[f + 3] + ss[FD + f + 3] };
  *(float4*)(xo + (size_t)n * FD + f) = o;
}

// ---------------------------------------------------------------- pooling
__global__ void pool_count_kernel(const int* __restrict__ batch, float* __restrict__ counts, int N) {
  int t = blockIdx.x * blockDim.x + threadIdx.x;
  if (t < N) atomicAdd(&counts[batch[t]], 1.0f);
}

__global__ void pool_accum_kernel(const float* __restrict__ X, const int* __restrict__ batch,
                                  float* __restrict__ pool, int N) {
  int t = blockIdx.x * blockDim.x + threadIdx.x;
  int n = t >> 5;
  if (n >= N) return;
  int f = (t & 31) * 4;
  int b = batch[n];
  float4 x = *(const float4*)(X + (size_t)n * FD + f);
  float* o = pool + (size_t)b * FD + f;
  atomicAdd(o + 0, x.x);
  atomicAdd(o + 1, x.y);
  atomicAdd(o + 2, x.z);
  atomicAdd(o + 3, x.w);
}

__global__ void pool_div_kernel(float* __restrict__ pool, const float* __restrict__ counts, int BD) {
  int t = blockIdx.x * blockDim.x + threadIdx.x;
  if (t >= BD) return;
  float c = counts[t >> 7];
  pool[t] /= fmaxf(c, 1.0f);
}

// ---------------------------------------------------------------- driver
extern "C" void kernel_launch(void* const* d_in, const int* in_sizes, int n_in,
                              void* d_out, int out_size, void* d_ws, size_t ws_size,
                              hipStream_t stream) {
  const float* x_in   = (const float*)d_in[0];
  const int*   ei     = (const int*)d_in[1];
  const int*   batch  = (const int*)d_in[2];
  const float* Ws     = (const float*)d_in[3];
  const float* bs     = (const float*)d_in[4];
  const float* gammas = (const float*)d_in[5];
  const float* betas  = (const float*)d_in[6];

  const int N = in_sizes[0] / FD;
  const int E = in_sizes[1] / 2;
  const int L = in_sizes[3] / (FD * FD);
  const int B = (int)(((long long)out_size - (long long)N * FD) / FD);

  const int* src = ei;
  const int* dst = ei + E;

  float* outPool = (float*)d_out;                       // [B, FD]
  float* outX    = (float*)d_out + (size_t)B * FD;      // [N, FD]

  // Workspace: h[N*FD] + agg[N*FD] + dinv[N] + sums[2*FD] + counts + Wh  (~103 MB)
  char* w = (char*)d_ws;
  float*    h    = (float*)w;  w += (size_t)N * FD * sizeof(float);
  float*    agg  = (float*)w;  w += (size_t)N * FD * sizeof(float);
  float*    dinv = (float*)w;  w += (size_t)((N + 15) & ~15) * sizeof(float);
  float*    sums = (float*)w;  w += 2 * FD * sizeof(float);
  float*    cnts = (float*)w;  w += (size_t)((B + 15) & ~15) * sizeof(float);
  _Float16* Wh   = (_Float16*)w;

  const int TB = 256;
  auto cdiv = [](long long a, long long b) { return (int)((a + b - 1) / b); };

  // degree (deg starts at 1.0 for self loop) -> dinv in place
  fill_kernel<<<cdiv(N, TB), TB, 0, stream>>>(dinv, 1.0f, N);
  deg_count_kernel<<<cdiv(E, TB), TB, 0, stream>>>(dst, dinv, E);
  dinv_kernel<<<cdiv(N, TB), TB, 0, stream>>>(dinv, N);

  const float* cur = x_in;
  for (int i = 0; i < L; ++i) {
    // x_cur is consumed only by the GEMM, so every layer may write outX.
    packW_kernel<<<cdiv(FD * FD, TB), TB, 0, stream>>>(Ws + (size_t)i * FD * FD, Wh);
    gemm_wmma_kernel<<<cdiv(N, 32), 256, 0, stream>>>(cur, Wh, h, N);
    agg_init_kernel<<<cdiv((long long)N * 32, TB), TB, 0, stream>>>(h, dinv, bs + (size_t)i * FD, agg, N);
    scatter_kernel<<<cdiv((long long)E * 32, TB), TB, 0, stream>>>(src, dst, dinv, h, agg, E);
    fill_kernel<<<1, 2 * FD, 0, stream>>>(sums, 0.0f, 2 * FD);
    bn_stats_kernel<<<cdiv(N, BN_ROWS), FD, 0, stream>>>(agg, sums, N);
    bn_finalize_kernel<<<1, FD, 0, stream>>>(sums, gammas + (size_t)i * FD, betas + (size_t)i * FD, N);
    bn_apply_kernel<<<cdiv((long long)N * 32, TB), TB, 0, stream>>>(agg, sums, outX, N);
    cur = outX;
  }

  // mean pooling by batch
  fill_kernel<<<cdiv(B, TB), TB, 0, stream>>>(cnts, 0.0f, B);
  fill_kernel<<<cdiv((long long)B * FD, TB), TB, 0, stream>>>(outPool, 0.0f, B * FD);
  pool_count_kernel<<<cdiv(N, TB), TB, 0, stream>>>(batch, cnts, N);
  pool_accum_kernel<<<cdiv((long long)N * 32, TB), TB, 0, stream>>>(cur, batch, outPool, N);
  pool_div_kernel<<<cdiv((long long)B * FD, TB), TB, 0, stream>>>(outPool, cnts, B * FD);
}
